// HAMTLayer_13271448944695
// MI455X (gfx1250) — compile-verified, matrix operand order checked by reading, not run on hardware
//
#include <hip/hip_runtime.h>
#include <math.h>

// ---------------------------------------------------------------------------
// HAMT layer for MI455X (gfx1250, wave32).
//   Phase 1: bf16 WMMA GEMMs, weights pre-transposed to [N,K]; per-wave
//            32x64 tiles (2 A-frags x 4 B-frags = 8 independent WMMAs/k-step).
//   Phase 2: circulant binding (VALU) + key normalization.
//   Phase 3: sequential scan, 1 workgroup/batch, hcm (f32 + bf16 shadow) in
//            LDS; retrieval via WMMA; gating/output GEMVs on packed-bf16
//            weights (1 dword = 2 k-rows) with f32 accumulate.
//   Phase 4: residual + LayerNorm epilogue.
// ---------------------------------------------------------------------------

typedef __bf16 bf16;
typedef bf16  v16bf __attribute__((ext_vector_type(16)));
typedef bf16  v8bf  __attribute__((ext_vector_type(8)));
typedef float v8f   __attribute__((ext_vector_type(8)));

#define B_ 8
#define S_ 1024
#define H_ 1024
#define D_ 512
#define NS_ 16
#define G_ 1552   /* H + D + N */
#define OI_ 1536  /* H + D */

__device__ __forceinline__ unsigned short f2bf_u16(float f) {
    unsigned u = __builtin_bit_cast(unsigned, f);
    u += 0x7FFFu + ((u >> 16) & 1u);           // round-to-nearest-even
    return (unsigned short)(u >> 16);
}
__device__ __forceinline__ bf16 f2bf(float f) {
    unsigned short h = f2bf_u16(f);
    return __builtin_bit_cast(bf16, h);
}
__device__ __forceinline__ float gelu_exact(float x) {
    return 0.5f * x * (1.0f + erff(x * 0.70710678118654752f));
}

// ---------------------------------------------------------------------------
// fp32 -> bf16 (row-major, A-side operands)
// ---------------------------------------------------------------------------
__global__ void k_f32_to_bf16(const float* __restrict__ src, bf16* __restrict__ dst, int n) {
    int i = blockIdx.x * blockDim.x + threadIdx.x;
    if (i < n) dst[i] = f2bf(src[i]);
}

// ---------------------------------------------------------------------------
// fp32 [K,N] -> bf16 transposed [N,K]  (one-time weight prep, LDS 32x32 tile)
// ---------------------------------------------------------------------------
__global__ void k_w_to_bf16T(const float* __restrict__ W, bf16* __restrict__ Wt,
                             int K, int N) {
    __shared__ float tile[32][33];
    const int n0 = blockIdx.x * 32, k0 = blockIdx.y * 32;
    const int tx = threadIdx.x & 31, ty = threadIdx.x >> 5;   // 32 x 8
    for (int r = ty; r < 32; r += 8)
        tile[r][tx] = W[(size_t)(k0 + r) * N + n0 + tx];
    __syncthreads();
    for (int r = ty; r < 32; r += 8)
        Wt[(size_t)(n0 + r) * K + k0 + tx] = f2bf(tile[tx][r]);
}

// ---------------------------------------------------------------------------
// fp32 [K,N] -> packed bf16 row-pairs: Wp[kk*N+n] = {W[2kk][n], W[2kk+1][n]}
// ---------------------------------------------------------------------------
__global__ void k_pack_rows_bf16(const float* __restrict__ W, unsigned* __restrict__ Wp,
                                 int K, int N) {
    int idx = blockIdx.x * blockDim.x + threadIdx.x;
    int total = (K / 2) * N;
    if (idx < total) {
        int kk = idx / N, n = idx - kk * N;
        unsigned lo = f2bf_u16(W[(size_t)(2 * kk) * N + n]);
        unsigned hi = f2bf_u16(W[(size_t)(2 * kk + 1) * N + n]);
        Wp[idx] = lo | (hi << 16);
    }
}

// ---------------------------------------------------------------------------
// WMMA GEMM:  C[M,N] = A[M,K] @ W[K,N] + bias, W TRANSPOSED as Wt[N,K].
// M=8192, K=1024. Block = 128 threads (4 waves); block tile 128(M)x64(N);
// wave tile 32x64: 2 A-frags x 4 B-frags -> 8 independent WMMAs per k-step.
// ---------------------------------------------------------------------------
__device__ __forceinline__ v16bf load_frag(const bf16* __restrict__ p) {
    v8bf lo = *(const v8bf*)p;
    v8bf hi = *(const v8bf*)(p + 16);
    v16bf r;
#pragma unroll
    for (int i = 0; i < 8; ++i) { r[i] = lo[i]; r[i + 8] = hi[i]; }
    return r;
}

__global__ void __launch_bounds__(128)
k_gemm_wmma(const bf16* __restrict__ A, const bf16* __restrict__ Wt,
            const float* __restrict__ bias, float* __restrict__ outF,
            bf16* __restrict__ outB, int N, int act) {
    const int K = 1024;
    const int tid  = threadIdx.x;
    const int wave = tid >> 5;
    const int lane = tid & 31;
    const int half = lane >> 4;
    const int lcol = lane & 15;
    const int n0 = blockIdx.x * 64;
    const int m0 = blockIdx.y * 128 + wave * 32;

    // ISA fragment layout: lane = row(A)/col(B) (lane&15); K bases half*8, half*8+16
    const bf16* ap0 = A  + (size_t)(m0 + lcol) * K + half * 8;
    const bf16* ap1 = ap0 + (size_t)16 * K;
    const bf16* bp0 = Wt + (size_t)(n0 + lcol) * K + half * 8;

    v8f acc[2][4] = {};

    for (int k0 = 0; k0 < K; k0 += 32) {
        v16bf a0 = load_frag(ap0 + k0);
        v16bf a1 = load_frag(ap1 + k0);
        v16bf bfr[4];
#pragma unroll
        for (int nt = 0; nt < 4; ++nt)
            bfr[nt] = load_frag(bp0 + (size_t)nt * 16 * K + k0);
        if (k0 + 32 < K) {                       // WGP-scope prefetch of next slab
            __builtin_prefetch(ap0 + k0 + 32, 0, 3);
            __builtin_prefetch(ap1 + k0 + 32, 0, 3);
            __builtin_prefetch(bp0 + k0 + 32, 0, 3);
        }
#pragma unroll
        for (int nt = 0; nt < 4; ++nt) {
            acc[0][nt] = __builtin_amdgcn_wmma_f32_16x16x32_bf16(
                false, a0, false, bfr[nt], (short)0, acc[0][nt], false, false);
            acc[1][nt] = __builtin_amdgcn_wmma_f32_16x16x32_bf16(
                false, a1, false, bfr[nt], (short)0, acc[1][nt], false, false);
        }
    }

    // epilogue: bias (+GELU), write f32 and/or bf16
#pragma unroll
    for (int mi = 0; mi < 2; ++mi) {
#pragma unroll
        for (int nt = 0; nt < 4; ++nt) {
            int n = n0 + nt * 16 + lcol;
            float bv = bias[n];
#pragma unroll
            for (int i = 0; i < 8; ++i) {
                int m = m0 + mi * 16 + half * 8 + i;  // C layout: VGPR i -> row i (+8)
                float v = acc[mi][nt][i] + bv;
                if (act == 1) v = gelu_exact(v);
                if (outF) outF[(size_t)m * N + n] = v;
                if (outB) outB[(size_t)m * N + n] = f2bf(v);
            }
        }
    }
}

// ---------------------------------------------------------------------------
// keys row-normalization (rows of [8192, 512]) + bf16 copy
// ---------------------------------------------------------------------------
__global__ void k_normalize_keys(float* __restrict__ keys, bf16* __restrict__ keysb) {
    const int r = blockIdx.x;
    float* kp = keys + (size_t)r * D_;
    float ss = 0.f;
    for (int i = threadIdx.x; i < D_; i += 128) { float v = kp[i]; ss += v * v; }
#pragma unroll
    for (int m = 1; m < 32; m <<= 1) ss += __shfl_xor(ss, m, 32);
    __shared__ float red[4];
    if ((threadIdx.x & 31) == 0) red[threadIdx.x >> 5] = ss;
    __syncthreads();
    float inv = 1.0f / fmaxf(sqrtf(red[0] + red[1] + red[2] + red[3]), 1e-12f);
    for (int i = threadIdx.x; i < D_; i += 128) {
        float v = kp[i] * inv;
        kp[i] = v;
        keysb[(size_t)r * D_ + i] = f2bf(v);
    }
}

// ---------------------------------------------------------------------------
// bound = circular convolution of items[b,s,:] with pos_keys[s,:]
// ---------------------------------------------------------------------------
__global__ void k_bound_conv(const float* __restrict__ items, const float* __restrict__ pos,
                             float* __restrict__ bound) {
    __shared__ float it[D_], pk[D_];
    const int s = blockIdx.x, b = blockIdx.y;
    const size_t row = (size_t)b * S_ + s;
    for (int i = threadIdx.x; i < D_; i += 256) {
        it[i] = items[row * D_ + i];
        pk[i] = pos[(size_t)s * D_ + i];
    }
    __syncthreads();
    for (int d = threadIdx.x; d < D_; d += 256) {
        float acc = 0.f;
        for (int j = 0; j < D_; ++j) acc = fmaf(it[j], pk[(d - j) & (D_ - 1)], acc);
        bound[row * D_ + d] = acc;
    }
}

// ---------------------------------------------------------------------------
// Sequential HAMT scan. One workgroup (32 waves) per batch. hcm f32 + bf16
// shadow in LDS; retrieval A-frags = two aligned ds_load_b128; ctx reuses
// gin[H..H+D). Gating/output GEMVs read packed-bf16 weights (2 rows/dword).
// ---------------------------------------------------------------------------
__global__ void __launch_bounds__(1024)
k_hamt_scan(const float* __restrict__ queries, const float* __restrict__ items,
            const float* __restrict__ bound, const bf16* __restrict__ keysb,
            const float* __restrict__ Wsa, const float* __restrict__ bsa,
            const unsigned* __restrict__ Wg1p, const float* __restrict__ bg1,
            const float* __restrict__ Wg2, const float* __restrict__ bg2,
            const unsigned* __restrict__ Wop, const float* __restrict__ bo,
            float* __restrict__ outs, float* __restrict__ hcm_out) {
    __shared__ float hcm[NS_][D_];                         // 32 KB
    __shared__ __align__(16) bf16 hcmb[NS_][D_];           // 16 KB shadow
    __shared__ __align__(16) bf16 keyrow[D_];
    __shared__ float bnd[D_];
    __shared__ float gin[G_ + 16];                         // [q | items->ctx | stats]
    __shared__ float g1s[H_];
    __shared__ float scores[NS_], wts[NS_], gates[NS_];

    const int b    = blockIdx.x;
    const int tid  = threadIdx.x;
    const int wave = tid >> 5;
    const int lane = tid & 31;
    const int half = lane >> 4;
    const int lcol = lane & 15;

    for (int i = tid; i < NS_ * D_; i += 1024) {
        ((float*)hcm)[i] = 0.f;
        ((bf16*)hcmb)[i] = f2bf(0.f);
    }
    __syncthreads();

    for (int t = 0; t < S_; ++t) {
        const size_t row = (size_t)b * S_ + t;

        // -- phase 0: stage per-step vectors; stats into gin tail ----------
        for (int i = tid; i < H_; i += 1024) gin[i] = queries[row * H_ + i];
        for (int i = tid; i < D_; i += 1024) {
            gin[H_ + i] = items[row * D_ + i];
            bnd[i]      = bound[row * D_ + i];
            keyrow[i]   = keysb[row * D_ + i];
        }
        if (tid < NS_) scores[tid] = 0.f;
        if (wave < NS_) {                        // stats[n] = ||hcm[n]||
            float ss = 0.f;
            for (int j = lane; j < D_; j += 32) { float v = hcm[wave][j]; ss += v * v; }
#pragma unroll
            for (int m = 1; m < 32; m <<= 1) ss += __shfl_xor(ss, m, 32);
            if (lane == 0) gin[OI_ + wave] = sqrtf(ss);
        }
        __syncthreads();

        // -- phase 1: retrieval WMMA; wave w owns d-columns [w*16, w*16+16) -
        const int n0 = wave * 16;
        const int dcol = n0 + lcol;
        v8f c = {};
        for (int k0 = 0; k0 < D_; k0 += 32) {
            v16bf af, bfrg;
            {   // A = hcm shadow, row = slot = lcol; two aligned 16B LDS loads
                v8bf alo = *(const v8bf*)&hcmb[lcol][k0 + half * 8];
                v8bf ahi = *(const v8bf*)&hcmb[lcol][k0 + half * 8 + 16];
#pragma unroll
                for (int i = 0; i < 8; ++i) { af[i] = alo[i]; af[i + 8] = ahi[i]; }
            }
#pragma unroll
            for (int i = 0; i < 8; ++i) {        // B = circulant of key row
                int k1 = k0 + half * 8 + i;
                bfrg[i]     = keyrow[(k1 - dcol) & (D_ - 1)];
                bfrg[i + 8] = keyrow[(k1 + 16 - dcol) & (D_ - 1)];
            }
            c = __builtin_amdgcn_wmma_f32_16x16x32_bf16(
                false, af, false, bfrg, (short)0, c, false, false);
        }
        // partial scores: score[m] += sum_d retrieved[m,d] * Wsa[d]
        {
            float wsa_l = Wsa[dcol];
#pragma unroll
            for (int i = 0; i < 8; ++i) {
                float v = c[i] * wsa_l;
#pragma unroll
                for (int m = 1; m < 16; m <<= 1) v += __shfl_xor(v, m, 32);
                if (lcol == 0) atomicAdd(&scores[i + half * 8], v);
            }
        }
        __syncthreads();

        // -- phase 2: softmax over N=16 ------------------------------------
        if (tid == 0) {
            float bias0 = bsa[0], mx = -1e30f;
            for (int n = 0; n < NS_; ++n) mx = fmaxf(mx, scores[n] + bias0);
            float sm = 0.f, e[NS_];
            for (int n = 0; n < NS_; ++n) { e[n] = __expf(scores[n] + bias0 - mx); sm += e[n]; }
            float inv = 1.0f / sm;
            for (int n = 0; n < NS_; ++n) wts[n] = e[n] * inv;
        }
        __syncthreads();

        // -- phase 3: gating GEMV on packed bf16 rows ----------------------
        {
            float acc = bg1[tid];                // thread tid owns output col tid
            for (int kk = 0; kk < G_ / 2; ++kk) {
                unsigned w2 = Wg1p[(size_t)kk * H_ + tid];
                float w0 = __builtin_bit_cast(float, w2 << 16);
                float w1 = __builtin_bit_cast(float, w2 & 0xFFFF0000u);
                acc = fmaf(gin[2 * kk], w0, fmaf(gin[2 * kk + 1], w1, acc));
            }
            g1s[tid] = gelu_exact(acc);
        }
        __syncthreads();

        // -- phase 4: ctx overwrites items span of gin; g2 -> gates --------
        {
            float v = 0.f;
#pragma unroll
            for (int i = 0; i < 8; ++i) v += c[i] * wts[i + half * 8];
            v += __shfl_xor(v, 16, 32);          // merge the two half-wave M ranges
            if (half == 0) gin[H_ + dcol] = v;   // gin[0:1536] becomes [q | ctx]
        }
        if (wave < NS_) {
            float acc = 0.f;
            for (int k = lane; k < H_; k += 32)
                acc = fmaf(g1s[k], Wg2[(size_t)k * NS_ + wave], acc);
#pragma unroll
            for (int m = 1; m < 32; m <<= 1) acc += __shfl_xor(acc, m, 32);
            if (lane == 0) gates[wave] = 1.0f / (1.0f + __expf(-(acc + bg2[wave])));
        }
        __syncthreads();

        // -- phase 5: memory update (+bf16 shadow) and output GEMV ---------
        for (int i = tid; i < NS_ * D_; i += 1024) {
            int n = i >> 9, d = i & (D_ - 1);
            float nv = 0.9f * hcm[n][d] + gates[n] * bnd[d];
            hcm[n][d]  = nv;
            hcmb[n][d] = f2bf(nv);
        }
        {
            float acc = bo[tid];
            for (int kk = 0; kk < OI_ / 2; ++kk) {
                unsigned w2 = Wop[(size_t)kk * H_ + tid];
                float w0 = __builtin_bit_cast(float, w2 << 16);
                float w1 = __builtin_bit_cast(float, w2 & 0xFFFF0000u);
                acc = fmaf(gin[2 * kk], w0, fmaf(gin[2 * kk + 1], w1, acc));
            }
            outs[row * H_ + tid] = acc;
        }
        __syncthreads();
    }

    for (int i = tid; i < NS_ * D_; i += 1024)
        hcm_out[(size_t)b * NS_ * D_ + i] = ((float*)hcm)[i];
}

// ---------------------------------------------------------------------------
// residual + LayerNorm epilogue
// ---------------------------------------------------------------------------
__global__ void k_final_ln(const float* __restrict__ hs, const float* __restrict__ outs,
                           const float* __restrict__ g, const float* __restrict__ be,
                           float* __restrict__ out) {
    const int r = blockIdx.x;
    __shared__ float xs[H_];
    __shared__ float rs[8], rq[8];
    float lsum = 0.f, lsq = 0.f;
    for (int i = threadIdx.x; i < H_; i += 256) {
        float v = hs[(size_t)r * H_ + i] + outs[(size_t)r * H_ + i];
        xs[i] = v; lsum += v; lsq += v * v;
    }
#pragma unroll
    for (int m = 1; m < 32; m <<= 1) { lsum += __shfl_xor(lsum, m, 32); lsq += __shfl_xor(lsq, m, 32); }
    if ((threadIdx.x & 31) == 0) { rs[threadIdx.x >> 5] = lsum; rq[threadIdx.x >> 5] = lsq; }
    __syncthreads();
    float sum = 0.f, sq = 0.f;
    for (int i = 0; i < 8; ++i) { sum += rs[i]; sq += rq[i]; }
    float mu  = sum * (1.0f / H_);
    float var = sq * (1.0f / H_) - mu * mu;
    float inv = rsqrtf(var + 1e-5f);
    for (int i = threadIdx.x; i < H_; i += 256)
        out[(size_t)r * H_ + i] = (xs[i] - mu) * inv * g[i] + be[i];
}

// ---------------------------------------------------------------------------
// host launcher
// ---------------------------------------------------------------------------
extern "C" void kernel_launch(void* const* d_in, const int* in_sizes, int n_in,
                              void* d_out, int out_size, void* d_ws, size_t ws_size,
                              hipStream_t stream) {
    const float* hs   = (const float*)d_in[0];
    const float* pos  = (const float*)d_in[1];
    const float* Wi   = (const float*)d_in[2];
    const float* bi   = (const float*)d_in[3];
    const float* Wq   = (const float*)d_in[4];
    const float* bq   = (const float*)d_in[5];
    const float* Wk1  = (const float*)d_in[6];
    const float* bk1  = (const float*)d_in[7];
    const float* Wk2  = (const float*)d_in[8];
    const float* bk2  = (const float*)d_in[9];
    const float* Wsa  = (const float*)d_in[10];
    const float* bsa  = (const float*)d_in[11];
    const float* Wg1  = (const float*)d_in[12];
    const float* bg1  = (const float*)d_in[13];
    const float* Wg2  = (const float*)d_in[14];
    const float* bg2  = (const float*)d_in[15];
    const float* Wo   = (const float*)d_in[16];
    const float* bo   = (const float*)d_in[17];
    const float* lng  = (const float*)d_in[18];
    const float* lnb  = (const float*)d_in[19];

    float* out     = (float*)d_out;
    float* hcm_out = out + (size_t)B_ * S_ * H_;   // [8,16,512] after [8,1024,1024]

    // ---- workspace carve (256B aligned) ----------------------------------
    char* ws = (char*)d_ws;
    size_t off = 0;
    auto carve = [&](size_t bytes) -> char* {
        char* p = ws + off;
        off = (off + bytes + 255) & ~(size_t)255;
        return p;
    };
    const size_t MT = (size_t)B_ * S_;             // 8192 rows
    bf16*     hsb     = (bf16*)carve(MT * H_ * 2);
    bf16*     WiT     = (bf16*)carve((size_t)H_ * D_ * 2);   // [512,1024]
    bf16*     WqT     = (bf16*)carve((size_t)H_ * H_ * 2);   // [1024,1024]
    bf16*     Wk1T    = (bf16*)carve((size_t)H_ * H_ * 2);   // [1024,1024]
    bf16*     Wk2T    = (bf16*)carve((size_t)H_ * D_ * 2);   // [512,1024]
    float*    items   = (float*)carve(MT * D_ * 4);
    float*    queries = (float*)carve(MT * H_ * 4);
    bf16*     qb      = (bf16*)carve(MT * H_ * 2);
    bf16*     k1b     = (bf16*)carve(MT * H_ * 2);
    float*    keys    = (float*)carve(MT * D_ * 4);
    bf16*     keysb   = (bf16*)carve(MT * D_ * 2);
    float*    bound   = (float*)carve(MT * D_ * 4);
    float*    outsbuf = (float*)carve(MT * H_ * 4);
    unsigned* Wg1p    = (unsigned*)carve((size_t)(G_ / 2) * H_ * 4);
    unsigned* Wop     = (unsigned*)carve((size_t)(OI_ / 2) * H_ * 4);

    // ---- conversions / weight prep ---------------------------------------
    {
        int n = (int)(MT * H_);
        k_f32_to_bf16<<<(n + 255) / 256, 256, 0, stream>>>(hs, hsb, n);
    }
    k_w_to_bf16T<<<dim3(D_ / 32, H_ / 32), 256, 0, stream>>>(Wi,  WiT,  H_, D_);
    k_w_to_bf16T<<<dim3(H_ / 32, H_ / 32), 256, 0, stream>>>(Wq,  WqT,  H_, H_);
    k_w_to_bf16T<<<dim3(H_ / 32, H_ / 32), 256, 0, stream>>>(Wk1, Wk1T, H_, H_);
    k_w_to_bf16T<<<dim3(D_ / 32, H_ / 32), 256, 0, stream>>>(Wk2, Wk2T, H_, D_);
    {
        int n1 = (G_ / 2) * H_;
        k_pack_rows_bf16<<<(n1 + 255) / 256, 256, 0, stream>>>(Wg1, Wg1p, G_, H_);
        int n2 = (OI_ / 2) * H_;
        k_pack_rows_bf16<<<(n2 + 255) / 256, 256, 0, stream>>>(Wo, Wop, OI_, H_);
    }

    dim3 blk(128);
    // items = hs @ Wi + bi
    k_gemm_wmma<<<dim3(D_ / 64, (int)(MT / 128)), blk, 0, stream>>>(
        hsb, WiT, bi, items, nullptr, D_, 0);
    // queries = hs @ Wq + bq  (f32 for scan, bf16 for next GEMM)
    k_gemm_wmma<<<dim3(H_ / 64, (int)(MT / 128)), blk, 0, stream>>>(
        hsb, WqT, bq, queries, qb, H_, 0);
    // k1 = gelu(queries @ Wk1 + bk1)   (bf16 only)
    k_gemm_wmma<<<dim3(H_ / 64, (int)(MT / 128)), blk, 0, stream>>>(
        qb, Wk1T, bk1, nullptr, k1b, H_, 1);
    // keys = k1 @ Wk2 + bk2
    k_gemm_wmma<<<dim3(D_ / 64, (int)(MT / 128)), blk, 0, stream>>>(
        k1b, Wk2T, bk2, keys, nullptr, D_, 0);

    k_normalize_keys<<<(int)MT, 128, 0, stream>>>(keys, keysb);
    k_bound_conv<<<dim3(S_, B_), 256, 0, stream>>>(items, pos, bound);

    k_hamt_scan<<<B_, 1024, 0, stream>>>(queries, items, bound, keysb,
                                         Wsa, bsa, Wg1p, bg1, Wg2, bg2, Wop, bo,
                                         outsbuf, hcm_out);

    k_final_ln<<<(int)MT, 256, 0, stream>>>(hs, outsbuf, lng, lnb, out);
    (void)in_sizes; (void)n_in; (void)out_size; (void)ws_size;
}
